// Transformer_63256278335575
// MI455X (gfx1250) — matrix operand order repaired and checked by reading, not compiled, not run on hardware
//
#include <hip/hip_runtime.h>
#include <hip/hip_bf16.h>

// CDNA5 (gfx1250) wave32 WMMA transformer encoder.
typedef __attribute__((ext_vector_type(16))) __bf16 v16bf;
typedef __attribute__((ext_vector_type(8)))  __bf16 v8bf;
typedef __attribute__((ext_vector_type(8)))  float  v8f;

#define LAYERS 3
#define DMODEL 256
#define HN     2
#define DHEAD  128
#define FFDIM  1024
#define BATCH  16
#define SEQ    1024
#define M_TOT  (BATCH * SEQ)   // 16384 rows

__device__ __forceinline__ __bf16 f2bf(float f) {
  unsigned u = __builtin_bit_cast(unsigned, f);
  u += 0x7fffu + ((u >> 16) & 1u);                  // round-to-nearest-even
  unsigned short s = (unsigned short)(u >> 16);
  return __builtin_bit_cast(__bf16, s);
}
__device__ __forceinline__ unsigned short bf2u(__bf16 b) {
  return __builtin_bit_cast(unsigned short, b);
}

// ---------------------------------------------------------------------------
// Weight pre-pack: fp32 W[K x N] -> bf16 in this file's B-fragment order (contiguous k, matching its A loads).
// Packed index for (k, n):
//   chunk = k>>5, g = (k>>4)&1, j = k&15, ct = n>>4, c = n&15, lane = c+16g
//   idx = ((chunk*(N/16) + ct)*32 + lane)*16 + j
// so a B-fragment is one contiguous 32B run per lane.
// ---------------------------------------------------------------------------
__global__ __launch_bounds__(256) void pack_w(const float* __restrict__ W,
                                              unsigned short* __restrict__ out,
                                              int K, int N) {
  const int tid = blockIdx.x * 256 + threadIdx.x;
  if (tid >= K * N) return;
  const int k = tid / N, n = tid % N;
  const int chunk = k >> 5, g = (k >> 4) & 1, j = k & 15;
  const int ct = n >> 4, c = n & 15, lane = c + 16 * g;
  const size_t idx = (((size_t)chunk * (N >> 4) + ct) * 32 + lane) * 16 + j;
  out[idx] = bf2u(f2bf(W[tid]));
}

// ---------------------------------------------------------------------------
// GEMM: C[M x N] = act(A[M x K] @ W[K x N] + bias), packed bf16 W.
// Wave tile 16 rows x 64 cols (4 accumulators, A reused 4x).
// QKV_SPLIT: columns >= 512 (the V block) are stored transposed into vt
//            [dim][M_TOT] so attention V fragments are contiguous.
// MIRROR: also emit a bf16 copy of the fp32 output (embedding -> hbf).
// ---------------------------------------------------------------------------
template <bool A_BF16, bool RELU, bool OUT_BF16, bool QKV_SPLIT, bool MIRROR>
__global__ __launch_bounds__(256) void gemm_wmma(
    const void* __restrict__ Ap, int lda,
    const unsigned short* __restrict__ Wp, const float* __restrict__ bias,
    void* __restrict__ Op, int ldo, int N, int K,
    unsigned short* __restrict__ mirror, unsigned short* __restrict__ vt)
{
  const int lane   = threadIdx.x & 31;
  const int wave   = threadIdx.x >> 5;
  const int waveId = blockIdx.x * 8 + wave;
  const int nColT  = N >> 6;
  const int row0   = (waveId / nColT) * 16;
  const int col0   = (waveId % nColT) * 64;
  const int r = lane & 15, g = lane >> 4;
  const int nCt = N >> 4;

  v8f acc[4] = {};
  for (int k0 = 0; k0 < K; k0 += 32) {
    v16bf a;
    if constexpr (A_BF16) {
      const v8bf* p = (const v8bf*)((const unsigned short*)Ap +
                                    (size_t)(row0 + r) * lda + k0 + 16 * g);
      const v8bf lo = p[0], hi = p[1];
#pragma unroll
      for (int i = 0; i < 8; ++i) { a[i] = lo[i]; a[i + 8] = hi[i]; }
    } else {
      const float* p = (const float*)Ap + (size_t)(row0 + r) * lda + k0 + 16 * g;
#pragma unroll
      for (int i = 0; i < 8; ++i) { a[i] = f2bf(p[i]); a[i + 8] = f2bf(p[8 + i]); }
    }
    const unsigned short* wbase =
        Wp + (size_t)(k0 >> 5) * nCt * 512 + (size_t)lane * 16;
#pragma unroll
    for (int t = 0; t < 4; ++t) {
      const v16bf bfr = *(const v16bf*)(wbase + (size_t)((col0 >> 4) + t) * 512);
      acc[t] = __builtin_amdgcn_wmma_f32_16x16x32_bf16(
          false, a, false, bfr, (short)0, acc[t], false, false);
    }
  }
#pragma unroll
  for (int t = 0; t < 4; ++t) {
    const int c = col0 + 16 * t + r;
    const float bz = bias[c];
#pragma unroll
    for (int v = 0; v < 8; ++v) {
      float val = acc[t][v] + bz;
      if constexpr (RELU) val = fmaxf(val, 0.f);
      const int row = row0 + v + 8 * g;
      if constexpr (QKV_SPLIT) {
        if (c >= 512) {                       // V block -> transposed store
          vt[(size_t)(c - 512) * M_TOT + row] = bf2u(f2bf(val));
          continue;
        }
      }
      const size_t idx = (size_t)row * ldo + c;
      if constexpr (OUT_BF16) {
        ((unsigned short*)Op)[idx] = bf2u(f2bf(val));
      } else {
        ((float*)Op)[idx] = val;
        if constexpr (MIRROR) mirror[idx] = bf2u(f2bf(val));
      }
    }
  }
}

// ---------------------------------------------------------------------------
// Residual + LayerNorm over D=256: one wave per row, shuffle reduction.
// Writes fp32 (residual stream) and a bf16 mirror (GEMM A-side feed).
// ---------------------------------------------------------------------------
__global__ __launch_bounds__(256) void ln_res(
    const float* __restrict__ base, const float* __restrict__ delta,
    const float* __restrict__ gamma, const float* __restrict__ beta,
    float* __restrict__ out, unsigned short* __restrict__ bfout)
{
  const int lane = threadIdx.x & 31;
  const int wave = threadIdx.x >> 5;
  const size_t row = (size_t)blockIdx.x * 8 + wave;
  const float* hb = base + row * DMODEL;
  const float* db = delta + row * DMODEL;
  float x[8], s = 0.f, s2 = 0.f;
#pragma unroll
  for (int i = 0; i < 8; ++i) {
    float v = hb[lane + 32 * i] + db[lane + 32 * i];
    x[i] = v; s += v; s2 += v * v;
  }
#pragma unroll
  for (int m = 1; m < 32; m <<= 1) {
    s  += __shfl_xor(s,  m, 32);
    s2 += __shfl_xor(s2, m, 32);
  }
  const float mean = s * (1.f / 256.f);
  const float var  = s2 * (1.f / 256.f) - mean * mean;
  const float rs   = __frsqrt_rn(var + 1e-5f);
  float* ob = out + row * DMODEL;
  unsigned short* bb = bfout + row * DMODEL;
#pragma unroll
  for (int i = 0; i < 8; ++i) {
    const int c = lane + 32 * i;
    const float y = (x[i] - mean) * rs * gamma[c] + beta[c];
    ob[c] = y;
    bb[c] = bf2u(f2bf(y));
  }
}

// ---------------------------------------------------------------------------
// Fused masked attention (flash-style, online softmax).
// One wave per (b, h, 16-query tile). qkv bf16 [B*N, 768] holds Q|K
// (cols [0,512)); V lives transposed in vt [256][B*N]. All fragment loads
// are contiguous 32B per lane. P is relaid out through 1KB LDS with an
// explicit s_wait_dscnt.
// ---------------------------------------------------------------------------
__global__ __launch_bounds__(32) void attn_wmma(
    const unsigned short* __restrict__ qkv,
    const unsigned short* __restrict__ vt,
    const int* __restrict__ valid,
    unsigned short* __restrict__ out)
{
  __shared__ __bf16 P[16][32];
  const int lane = threadIdx.x;
  const int bid  = blockIdx.x;              // b*(H*64) + h*64 + qt
  const int qt = bid & 63;
  const int hh = (bid >> 6) & 1;
  const int b  = bid >> 7;
  const int r = lane & 15, g = lane >> 4;
  const int vn = valid[b];
  const size_t base = (size_t)b * SEQ * 768;
  const int q0 = qt * 16;

  // Preload Q A-fragments: 16 queries x 128 dims = 4 fragments.
  v16bf qf[4];
#pragma unroll
  for (int kc = 0; kc < 4; ++kc) {
    const v8bf* p = (const v8bf*)(qkv + base + (size_t)(q0 + r) * 768 +
                                  hh * 128 + kc * 32 + 16 * g);
    const v8bf lo = p[0], hi = p[1];
#pragma unroll
    for (int i = 0; i < 8; ++i) { qf[kc][i] = lo[i]; qf[kc][i + 8] = hi[i]; }
  }

  float mrun[8], lrun[8];
  v8f o[8] = {};
#pragma unroll
  for (int v = 0; v < 8; ++v) { mrun[v] = -1e30f; lrun[v] = 0.f; }
  const float scale = 0.0883883476483184f;  // 1/sqrt(128)
  const size_t vtrow = (size_t)b * SEQ;

#pragma unroll 1
  for (int j0 = 0; j0 < SEQ; j0 += 32) {
    // Prefetch next key tile (speculative, lowers to global_prefetch_b8).
    __builtin_prefetch(qkv + base + (size_t)(j0 + 32 + r) * 768 + 256 + hh * 128, 0, 1);

    // S tiles: keys [j0, j0+16) and [j0+16, j0+32); K frags are contiguous.
    v8f s0 = {}, s1 = {};
#pragma unroll
    for (int kc = 0; kc < 4; ++kc) {
      const unsigned short* kp = qkv + base + (size_t)(j0 + r) * 768 + 256 +
                                 hh * 128 + kc * 32 + 16 * g;
      const v16bf kb0 = *(const v16bf*)kp;
      const v16bf kb1 = *(const v16bf*)(kp + (size_t)16 * 768);
      s0 = __builtin_amdgcn_wmma_f32_16x16x32_bf16(false, qf[kc], false, kb0,
                                                   (short)0, s0, false, false);
      s1 = __builtin_amdgcn_wmma_f32_16x16x32_bf16(false, qf[kc], false, kb1,
                                                   (short)0, s1, false, false);
    }
    // Mask + scale; online softmax stats per query row (index v+8g).
    const bool k0inv = (j0 + r) >= vn;
    const bool k1inv = (j0 + 16 + r) >= vn;
    float mnew[8], alpha[8], p0v[8], p1v[8], rsum[8];
#pragma unroll
    for (int v = 0; v < 8; ++v) {
      const bool qinv = (q0 + v + 8 * g) >= vn;
      s0[v] = (qinv || k0inv) ? -1e9f : s0[v] * scale;
      s1[v] = (qinv || k1inv) ? -1e9f : s1[v] * scale;
      mnew[v] = fmaxf(s0[v], s1[v]);
    }
#pragma unroll
    for (int m = 1; m < 16; m <<= 1)
#pragma unroll
      for (int v = 0; v < 8; ++v)
        mnew[v] = fmaxf(mnew[v], __shfl_xor(mnew[v], m, 32));
#pragma unroll
    for (int v = 0; v < 8; ++v) {
      mnew[v]  = fmaxf(mnew[v], mrun[v]);
      alpha[v] = __expf(mrun[v] - mnew[v]);
      mrun[v]  = mnew[v];
      p0v[v] = __expf(s0[v] - mnew[v]);
      p1v[v] = __expf(s1[v] - mnew[v]);
      rsum[v] = p0v[v] + p1v[v];
    }
#pragma unroll
    for (int m = 1; m < 16; m <<= 1)
#pragma unroll
      for (int v = 0; v < 8; ++v)
        rsum[v] += __shfl_xor(rsum[v], m, 32);
#pragma unroll
    for (int v = 0; v < 8; ++v) lrun[v] = lrun[v] * alpha[v] + rsum[v];
#pragma unroll
    for (int t = 0; t < 8; ++t)
#pragma unroll
      for (int v = 0; v < 8; ++v) o[t][v] *= alpha[v];

    // Relayout P (D-layout fp32 regs) -> A-fragment via LDS, as bf16.
#pragma unroll
    for (int v = 0; v < 8; ++v) {
      P[v + 8 * g][r]      = f2bf(p0v[v]);
      P[v + 8 * g][16 + r] = f2bf(p1v[v]);
    }
    asm volatile("s_wait_dscnt 0" ::: "memory");
    v16bf pf;
    {
      const v8bf lo = *(const v8bf*)&P[r][16 * g];
      const v8bf hi = *(const v8bf*)&P[r][16 * g + 8];
#pragma unroll
      for (int i = 0; i < 8; ++i) { pf[i] = lo[i]; pf[i + 8] = hi[i]; }
    }
    asm volatile("s_wait_dscnt 0" ::: "memory");

    // O[16 x 128] += P[16 x 32] * V[32 x 128]; V frags contiguous from vt.
#pragma unroll
    for (int t = 0; t < 8; ++t) {
      const v16bf vb = *(const v16bf*)(vt +
          (size_t)(hh * 128 + t * 16 + r) * M_TOT + vtrow + j0 + 16 * g);
      o[t] = __builtin_amdgcn_wmma_f32_16x16x32_bf16(false, pf, false, vb,
                                                     (short)0, o[t], false, false);
    }
  }

  // Normalize and store (bf16, consumed by proj GEMM).
#pragma unroll
  for (int t = 0; t < 8; ++t)
#pragma unroll
    for (int v = 0; v < 8; ++v) {
      const float val = o[t][v] / lrun[v];
      const size_t idx =
          ((size_t)b * SEQ + q0 + v + 8 * g) * 256 + hh * 128 + t * 16 + r;
      out[idx] = bf2u(f2bf(val));
    }
}

// ---------------------------------------------------------------------------
extern "C" void kernel_launch(void* const* d_in, const int* in_sizes, int n_in,
                              void* d_out, int out_size, void* d_ws, size_t ws_size,
                              hipStream_t stream) {
  (void)in_sizes; (void)n_in; (void)out_size; (void)ws_size;
  const float* x      = (const float*)d_in[0];
  const int*   valid  = (const int*)d_in[1];
  const float* emb_W  = (const float*)d_in[2];
  const float* emb_b  = (const float*)d_in[3];
  const float* pre_W  = (const float*)d_in[4];
  const float* pre_b  = (const float*)d_in[5];
  const float* proj_W = (const float*)d_in[6];
  const float* proj_b = (const float*)d_in[7];
  const float* ln1_g  = (const float*)d_in[8];
  const float* ln1_b  = (const float*)d_in[9];
  const float* mlp_W1 = (const float*)d_in[10];
  const float* mlp_b1 = (const float*)d_in[11];
  const float* mlp_W2 = (const float*)d_in[12];
  const float* mlp_b2 = (const float*)d_in[13];
  const float* ln2_g  = (const float*)d_in[14];
  const float* ln2_b  = (const float*)d_in[15];

  // Workspace layout (~117 MB)
  char* ws = (char*)d_ws;
  float*          h    = (float*)(ws);                                // 16 MB f32
  float*          tmp  = (float*)(ws + (size_t)(16u << 20));          // 16 MB f32
  unsigned short* qkv  = (unsigned short*)(ws + (size_t)(32u << 20)); // 24 MB bf16 (Q|K)
  unsigned short* attn = (unsigned short*)(ws + (size_t)(56u << 20)); //  8 MB bf16
  unsigned short* ff   = (unsigned short*)(ws + (size_t)(64u << 20)); // 32 MB bf16
  unsigned short* vt   = (unsigned short*)(ws + (size_t)(96u << 20)); //  8 MB bf16 V^T
  unsigned short* hbf  = (unsigned short*)(ws + (size_t)(104u << 20));//  8 MB bf16
  unsigned short* wp   = (unsigned short*)(ws + (size_t)(112u << 20));// ~4.9 MB packed W

  unsigned short* wp_emb  = wp;                        // 256x256
  unsigned short* wp_pre  = wp + 65536;                // 3 x 256x768
  unsigned short* wp_proj = wp + 655360;               // 3 x 256x256
  unsigned short* wp_m1   = wp + 851968;               // 3 x 256x1024
  unsigned short* wp_m2   = wp + 1638400;              // 3 x 1024x256

  const dim3 blk(256);
  const int rowT = M_TOT / 16;   // 1024 row tiles

  // ---- one-shot weight packing (tiny; fully L2 resident afterwards) ----
  pack_w<<<dim3((256 * 256 + 255) / 256), blk, 0, stream>>>(emb_W, wp_emb, 256, 256);
  for (int i = 0; i < LAYERS; ++i) {
    pack_w<<<dim3((256 * 768 + 255) / 256), blk, 0, stream>>>(
        pre_W + (size_t)i * 256 * 768, wp_pre + (size_t)i * 196608, 256, 768);
    pack_w<<<dim3((256 * 256 + 255) / 256), blk, 0, stream>>>(
        proj_W + (size_t)i * 256 * 256, wp_proj + (size_t)i * 65536, 256, 256);
    pack_w<<<dim3((256 * 1024 + 255) / 256), blk, 0, stream>>>(
        mlp_W1 + (size_t)i * 256 * 1024, wp_m1 + (size_t)i * 262144, 256, 1024);
    pack_w<<<dim3((1024 * 256 + 255) / 256), blk, 0, stream>>>(
        mlp_W2 + (size_t)i * 1024 * 256, wp_m2 + (size_t)i * 262144, 1024, 256);
  }

  // Embedding: h = relu(x @ emb_W + emb_b), fp32 + bf16 mirror.
  gemm_wmma<false, true, false, false, true>
      <<<dim3(rowT * (256 / 64) / 8), blk, 0, stream>>>(
          x, 256, wp_emb, emb_b, h, 256, 256, 256, hbf, nullptr);

  for (int i = 0; i < LAYERS; ++i) {
    // qkv = h @ pre_W[i] + pre_b[i]; Q|K row-major bf16, V transposed to vt.
    gemm_wmma<true, false, true, true, false>
        <<<dim3(rowT * (768 / 64) / 8), blk, 0, stream>>>(
            hbf, 256, wp_pre + (size_t)i * 196608, pre_b + i * 768,
            qkv, 768, 768, 256, nullptr, vt);
    // fused masked attention
    attn_wmma<<<dim3(BATCH * HN * (SEQ / 16)), dim3(32), 0, stream>>>(
        qkv, vt, valid, attn);
    // tmp = attn @ proj_W[i] + proj_b[i]
    gemm_wmma<true, false, false, false, false>
        <<<dim3(rowT * (256 / 64) / 8), blk, 0, stream>>>(
            attn, 256, wp_proj + (size_t)i * 65536, proj_b + i * 256,
            tmp, 256, 256, 256, nullptr, nullptr);
    // h = LN1(h + tmp) (+ bf16 mirror)
    ln_res<<<dim3(M_TOT / 8), blk, 0, stream>>>(
        h, tmp, ln1_g + i * 256, ln1_b + i * 256, h, hbf);
    // ff = relu(h @ mlp_W1[i] + b1)  (bf16)
    gemm_wmma<true, true, true, false, false>
        <<<dim3(rowT * (1024 / 64) / 8), blk, 0, stream>>>(
            hbf, 256, wp_m1 + (size_t)i * 262144, mlp_b1 + i * 1024,
            ff, 1024, 1024, 256, nullptr, nullptr);
    // tmp = relu(ff @ mlp_W2[i] + b2)
    gemm_wmma<true, true, false, false, false>
        <<<dim3(rowT * (256 / 64) / 8), blk, 0, stream>>>(
            ff, 1024, wp_m2 + (size_t)i * 262144, mlp_b2 + i * 256,
            tmp, 256, 256, 1024, nullptr, nullptr);
    // h = LN2(h + tmp); final layer writes straight to d_out
    float* lnout = (i == LAYERS - 1) ? (float*)d_out : h;
    ln_res<<<dim3(M_TOT / 8), blk, 0, stream>>>(
        h, tmp, ln2_g + i * 256, ln2_b + i * 256, lnout, hbf);
  }
}
